// EdgeAwareTransformerBackbone_41180146434511
// MI455X (gfx1250) — compile-verified, matrix operand order checked by reading, not run on hardware
//
#include <hip/hip_runtime.h>
#include <hip/hip_bf16.h>
#include <math.h>

// ---------------------------------------------------------------------------
// EdgeAwareTransformerBackbone for MI455X (gfx1250, wave32).
//  - Dense q/k/v/skip projections via v_wmma_f32_16x16x32_bf16 (fused 128x896 GEMM)
//  - Edge attention via wave-per-edge gather + scatter-softmax (f32 atomics into L2)
//  - ycat [N,896] f32 (~179MB) ~fits 192MB L2 -> random per-edge gathers are L2 hits
// ---------------------------------------------------------------------------

#define N_NODES 50000
#define N_EDGES 500000
#define HID     128
#define HC      256     // HEADS * HEAD_DIM
#define YW      896     // q(256)|k(256)|v(512..767)|skip(768..895)
#define LAYERS  3

typedef __attribute__((ext_vector_type(16))) __bf16 v16bf;
typedef __attribute__((ext_vector_type(8)))  float  v8f;

__device__ __forceinline__ unsigned short f2bf(float f) {
    unsigned int u = __float_as_uint(f);
    unsigned int r = (u + 0x7FFFu + ((u >> 16) & 1u)) >> 16;   // RNE
    return (unsigned short)r;
}

__device__ __forceinline__ void atomicMaxF32(float* addr, float val) {
    if (val >= 0.0f)
        atomicMax((int*)addr, __float_as_int(val));
    else
        atomicMin((unsigned int*)addr, (unsigned int)__float_as_int(val));
}

// ---------------- input layernorm: wave per node, 4 elems/lane --------------
__global__ void __launch_bounds__(256)
ln_in_kernel(const float* __restrict__ xin, const float* __restrict__ g,
             const float* __restrict__ b, float* __restrict__ xcur,
             unsigned short* __restrict__ xbf)
{
    int wid  = (blockIdx.x * blockDim.x + threadIdx.x) >> 5;
    int lane = threadIdx.x & 31;
    int nw   = (gridDim.x * blockDim.x) >> 5;
    for (int n = wid; n < N_NODES; n += nw) {
        const float* row = xin + (size_t)n * HID;
        float v[4]; float s = 0.f;
        #pragma unroll
        for (int j = 0; j < 4; ++j) { v[j] = row[lane * 4 + j]; s += v[j]; }
        #pragma unroll
        for (int m = 16; m >= 1; m >>= 1) s += __shfl_xor(s, m, 32);
        float mu = s * (1.f / HID);
        float var = 0.f;
        #pragma unroll
        for (int j = 0; j < 4; ++j) { float d = v[j] - mu; var += d * d; }
        #pragma unroll
        for (int m = 16; m >= 1; m >>= 1) var += __shfl_xor(var, m, 32);
        float r = rsqrtf(var * (1.f / HID) + 1e-5f);
        #pragma unroll
        for (int j = 0; j < 4; ++j) {
            int c = lane * 4 + j;
            float o = (v[j] - mu) * r * g[c] + b[c];
            xcur[(size_t)n * HID + c] = o;
            xbf [(size_t)n * HID + c] = f2bf(o);
        }
    }
}

// ------------- convert+transpose layer weights to bf16 [896][128] -----------
__global__ void __launch_bounds__(256)
wconv_kernel(const float* __restrict__ Wq, const float* __restrict__ Wk,
             const float* __restrict__ Wv, const float* __restrict__ Ws,
             unsigned short* __restrict__ Wt)
{
    int i = blockIdx.x * blockDim.x + threadIdx.x;
    if (i >= YW * HID) return;
    int col = i >> 7, k = i & 127;
    float w;
    if      (col < 256) w = Wq[k * HC + col];
    else if (col < 512) w = Wk[k * HC + (col - 256)];
    else if (col < 768) w = Wv[k * HC + (col - 512)];
    else                w = Ws[k * HID + (col - 768)];
    Wt[col * HID + k] = f2bf(w);
}

// -------- fused GEMM: ycat[N,896] = xbf[N,128] @ Wt^T + bias (WMMA bf16) ----
__global__ void __launch_bounds__(256)
gemm_kernel(const unsigned short* __restrict__ A,   // [N,128] bf16
            const unsigned short* __restrict__ Bt,  // [896,128] bf16
            const float* __restrict__ bq, const float* __restrict__ bk,
            const float* __restrict__ bv, const float* __restrict__ bs,
            float* __restrict__ Y)
{
    int wave = threadIdx.x >> 5;
    int lane = threadIdx.x & 31;
    int mtile = blockIdx.x;                    // 3125
    int ntile = blockIdx.y * 8 + wave;         // 56
    int l16   = lane & 15;
    int khalf = (lane >> 4) << 3;              // 0 or 8
    const unsigned short* ap = A  + ((size_t)(mtile * 16 + l16)) * HID + khalf;
    const unsigned short* bp = Bt + ((size_t)(ntile * 16 + l16)) * HID + khalf;

    union Frag { v16bf v; int4 q[2]; };
    v8f c = {};
    #pragma unroll
    for (int k0 = 0; k0 < HID; k0 += 32) {
        Frag a, b;
        a.q[0] = *(const int4*)(ap + k0);
        a.q[1] = *(const int4*)(ap + k0 + 16);
        b.q[0] = *(const int4*)(bp + k0);
        b.q[1] = *(const int4*)(bp + k0 + 16);
        c = __builtin_amdgcn_wmma_f32_16x16x32_bf16(
                false, a.v, false, b.v, (short)0, c, false, false);
    }

    int col  = ntile * 16 + l16;
    float bias = (col < 256) ? bq[col]
               : (col < 512) ? bk[col - 256]
               : (col < 768) ? bv[col - 512]
               :               bs[col - 768];
    int rbase = mtile * 16 + ((lane >> 4) << 3);
    #pragma unroll
    for (int gi = 0; gi < 8; ++gi)
        Y[(size_t)(rbase + gi) * YW + col] = c[gi] + bias;
}

// -------- per-layer init: out=0, m=-inf, denom=0 ----------------------------
__global__ void init_kernel(float* __restrict__ mbuf, float* __restrict__ denom,
                            float* __restrict__ out)
{
    int i = blockIdx.x * blockDim.x + threadIdx.x;
    if (i < N_NODES * HID) out[i] = 0.f;
    if (i < N_NODES * 2) { mbuf[i] = -INFINITY; denom[i] = 0.f; }
}

// -------- alpha: wave per edge, lanes 0-15 head0 / 16-31 head1 --------------
__global__ void __launch_bounds__(256)
alpha_kernel(const int* __restrict__ src, const int* __restrict__ tgt,
             const float* __restrict__ eattr, const float* __restrict__ We,
             const float* __restrict__ Y, float* __restrict__ alpha,
             float* __restrict__ mbuf)
{
    int wid  = (blockIdx.x * blockDim.x + threadIdx.x) >> 5;
    int lane = threadIdx.x & 31;
    int nw   = (gridDim.x * blockDim.x) >> 5;
    int h    = lane >> 4;
    int cb   = (lane & 15) * 8 + h * 128;      // offset within 256-wide q/k
    const float4* wep = (const float4*)(We + cb);
    float4 w0 = wep[0], w1 = wep[1];
    for (int e = wid; e < N_EDGES; e += nw) {
        int s = src[e], t = tgt[e];
        float ea = eattr[e];
        const float4* qp = (const float4*)(Y + (size_t)t * YW + cb);
        const float4* kp = (const float4*)(Y + (size_t)s * YW + 256 + cb);
        float4 q0 = qp[0], q1 = qp[1], k0 = kp[0], k1 = kp[1];
        float d = q0.x * (k0.x + ea * w0.x) + q0.y * (k0.y + ea * w0.y)
                + q0.z * (k0.z + ea * w0.z) + q0.w * (k0.w + ea * w0.w)
                + q1.x * (k1.x + ea * w1.x) + q1.y * (k1.y + ea * w1.y)
                + q1.z * (k1.z + ea * w1.z) + q1.w * (k1.w + ea * w1.w);
        #pragma unroll
        for (int m = 1; m < 16; m <<= 1) d += __shfl_xor(d, m, 32);
        d *= 0.08838834764831845f;             // 1/sqrt(128)
        if ((lane & 15) == 0) {
            alpha[(size_t)e * 2 + h] = d;
            atomicMaxF32(mbuf + (size_t)t * 2 + h, d);
        }
    }
}

// -------- exp(alpha - max) and denominators ---------------------------------
__global__ void __launch_bounds__(256)
expsum_kernel(const int* __restrict__ tgt, float* __restrict__ alpha,
              const float* __restrict__ mbuf, float* __restrict__ denom)
{
    int i = blockIdx.x * blockDim.x + threadIdx.x;
    if (i >= N_EDGES * 2) return;
    int e = i >> 1, h = i & 1;
    int t = tgt[e];
    float a = __expf(alpha[i] - mbuf[(size_t)t * 2 + h]);
    alpha[i] = a;
    atomicAdd(denom + (size_t)t * 2 + h, a);
}

// -------- message scatter (head-mean 0.5 folded into scale) -----------------
__global__ void __launch_bounds__(256)
msg_kernel(const int* __restrict__ src, const int* __restrict__ tgt,
           const float* __restrict__ eattr, const float* __restrict__ We,
           const float* __restrict__ Y, const float* __restrict__ alpha,
           const float* __restrict__ denom, float* __restrict__ out)
{
    int wid  = (blockIdx.x * blockDim.x + threadIdx.x) >> 5;
    int lane = threadIdx.x & 31;
    int nw   = (gridDim.x * blockDim.x) >> 5;
    int h    = lane >> 4;
    int cb   = (lane & 15) * 8 + h * 128;
    int cc   = (lane & 15) * 8;                // head-folded output column
    const float4* wep = (const float4*)(We + cb);
    float4 w0 = wep[0], w1 = wep[1];
    for (int e = wid; e < N_EDGES; e += nw) {
        int s = src[e], t = tgt[e];
        float ea = eattr[e];
        float dn = denom[(size_t)t * 2 + h];
        dn = (dn > 0.f) ? dn : 1.f;
        float a = alpha[(size_t)e * 2 + h] * 0.5f / dn;
        const float4* vp = (const float4*)(Y + (size_t)s * YW + 512 + cb);
        float4 v0 = vp[0], v1 = vp[1];
        float* op = out + (size_t)t * HID + cc;
        atomicAdd(op + 0, (v0.x + ea * w0.x) * a);
        atomicAdd(op + 1, (v0.y + ea * w0.y) * a);
        atomicAdd(op + 2, (v0.z + ea * w0.z) * a);
        atomicAdd(op + 3, (v0.w + ea * w0.w) * a);
        atomicAdd(op + 4, (v1.x + ea * w1.x) * a);
        atomicAdd(op + 5, (v1.y + ea * w1.y) * a);
        atomicAdd(op + 6, (v1.z + ea * w1.z) * a);
        atomicAdd(op + 7, (v1.w + ea * w1.w) * a);
    }
}

// -------- finalize: +skip, ELU, residual, layernorm -------------------------
__global__ void __launch_bounds__(256)
finalize_kernel(const float* __restrict__ out_acc, const float* __restrict__ Y,
                const float* __restrict__ g, const float* __restrict__ b,
                const float* __restrict__ xprev, float* __restrict__ dst,
                unsigned short* __restrict__ xbf, int write_bf)
{
    int wid  = (blockIdx.x * blockDim.x + threadIdx.x) >> 5;
    int lane = threadIdx.x & 31;
    int nw   = (gridDim.x * blockDim.x) >> 5;
    for (int n = wid; n < N_NODES; n += nw) {
        float v[4]; float s = 0.f;
        #pragma unroll
        for (int j = 0; j < 4; ++j) {
            int c = lane * 4 + j;
            float hv = out_acc[(size_t)n * HID + c] + Y[(size_t)n * YW + 768 + c];
            hv = (hv > 0.f) ? hv : (__expf(hv) - 1.f);         // ELU
            v[j] = xprev[(size_t)n * HID + c] + hv;             // residual
            s += v[j];
        }
        #pragma unroll
        for (int m = 16; m >= 1; m >>= 1) s += __shfl_xor(s, m, 32);
        float mu = s * (1.f / HID);
        float var = 0.f;
        #pragma unroll
        for (int j = 0; j < 4; ++j) { float d = v[j] - mu; var += d * d; }
        #pragma unroll
        for (int m = 16; m >= 1; m >>= 1) var += __shfl_xor(var, m, 32);
        float r = rsqrtf(var * (1.f / HID) + 1e-5f);
        #pragma unroll
        for (int j = 0; j < 4; ++j) {
            int c = lane * 4 + j;
            float o = (v[j] - mu) * r * g[c] + b[c];
            dst[(size_t)n * HID + c] = o;
            if (write_bf) xbf[(size_t)n * HID + c] = f2bf(o);
        }
    }
}

// ---------------------------------------------------------------------------
extern "C" void kernel_launch(void* const* d_in, const int* in_sizes, int n_in,
                              void* d_out, int out_size, void* d_ws, size_t ws_size,
                              hipStream_t stream)
{
    const float* x        = (const float*)d_in[0];
    const int*   eidx     = (const int*)  d_in[1];
    const float* eattr    = (const float*)d_in[2];
    const float* ln_in_g  = (const float*)d_in[3];
    const float* ln_in_b  = (const float*)d_in[4];
    const float* Wq       = (const float*)d_in[5];
    const float* bq       = (const float*)d_in[6];
    const float* Wk       = (const float*)d_in[7];
    const float* bk       = (const float*)d_in[8];
    const float* Wv       = (const float*)d_in[9];
    const float* bv       = (const float*)d_in[10];
    const float* We       = (const float*)d_in[11];
    const float* Wskip    = (const float*)d_in[12];
    const float* bskip    = (const float*)d_in[13];
    const float* ln_g     = (const float*)d_in[14];
    const float* ln_b     = (const float*)d_in[15];
    float* outp = (float*)d_out;

    const int* src = eidx;
    const int* tgt = eidx + N_EDGES;

    // ---- carve workspace (256B aligned) ----
    char* ws = (char*)d_ws;
    size_t off = 0;
    auto carve = [&](size_t bytes) {
        void* p = ws + off;
        off += (bytes + 255) & ~(size_t)255;
        return p;
    };
    float*          ycat  = (float*)         carve((size_t)N_NODES * YW  * 4); // 179.2MB
    float*          xcur  = (float*)         carve((size_t)N_NODES * HID * 4); //  25.6MB
    unsigned short* xbf   = (unsigned short*)carve((size_t)N_NODES * HID * 2); //  12.8MB
    unsigned short* Wt    = (unsigned short*)carve((size_t)YW * HID * 2);
    float*          alpha = (float*)         carve((size_t)N_EDGES * 2  * 4); //   4MB
    float*          mbuf  = (float*)         carve((size_t)N_NODES * 2  * 4);
    float*          denom = (float*)         carve((size_t)N_NODES * 2  * 4);
    float*          oacc  = (float*)         carve((size_t)N_NODES * HID * 4); //  25.6MB
    (void)ws_size; (void)in_sizes; (void)n_in; (void)out_size;

    // input layernorm
    ln_in_kernel<<<6250, 256, 0, stream>>>(x, ln_in_g, ln_in_b, xcur, xbf);

    for (int l = 0; l < LAYERS; ++l) {
        // weight convert/transpose to bf16
        wconv_kernel<<<(YW * HID + 255) / 256, 256, 0, stream>>>(
            Wq + (size_t)l * HID * HC, Wk + (size_t)l * HID * HC,
            Wv + (size_t)l * HID * HC, Wskip + (size_t)l * HID * HID, Wt);

        // fused q|k|v|skip projection via WMMA
        dim3 ggrid(N_NODES / 16, YW / 16 / 8);   // (3125, 7)
        gemm_kernel<<<ggrid, 256, 0, stream>>>(
            xbf, Wt, bq + (size_t)l * HC, bk + (size_t)l * HC,
            bv + (size_t)l * HC, bskip + (size_t)l * HID, ycat);

        // reset softmax state + accumulators
        init_kernel<<<(N_NODES * HID + 255) / 256, 256, 0, stream>>>(mbuf, denom, oacc);

        const float* Wel = We + (size_t)l * HC;
        alpha_kernel<<<(N_EDGES / 8 + 31) / 32 * 32 / 32 + 1, 256, 0, stream>>>(
            src, tgt, eattr, Wel, ycat, alpha, mbuf);
        expsum_kernel<<<(N_EDGES * 2 + 255) / 256, 256, 0, stream>>>(
            tgt, alpha, mbuf, denom);
        msg_kernel<<<(N_EDGES + 7) / 8, 256, 0, stream>>>(
            src, tgt, eattr, Wel, ycat, alpha, denom, oacc);

        // epilogue: skip + ELU + residual + LN
        int last = (l == LAYERS - 1);
        finalize_kernel<<<6250, 256, 0, stream>>>(
            oacc, ycat, ln_g + (size_t)l * HID, ln_b + (size_t)l * HID,
            xcur, last ? outp : xcur, xbf, last ? 0 : 1);
    }
}
